// DPLSTMModel_13056700580290
// MI455X (gfx1250) — compile-verified, matrix operand order checked by reading, not run on hardware
//
#include <hip/hip_runtime.h>
#include <hip/hip_bf16.h>

// ---------------------------------------------------------------------------
// LSTM LM forward for MI455X (gfx1250, wave32, WMMA + TDM).
// GEMMs: V_WMMA_F32_16X16X32_BF16 (bf16 in, f32 accumulate).
// A-operand tiles staged into LDS via TENSOR_LOAD_TO_LDS (TDM, TENSORcnt).
// ---------------------------------------------------------------------------

#define VOCAB 50257
#define EMB   1024
#define HID   1024
#define BATCH 8
#define SEQT  256
#define BT    (BATCH * SEQT)      // 2048
#define G4H   (4 * HID)           // 4096

typedef __bf16 v16bf __attribute__((ext_vector_type(16)));
typedef __bf16 v8bf  __attribute__((ext_vector_type(8)));
typedef float  v8f   __attribute__((ext_vector_type(8)));
typedef unsigned int v4u __attribute__((ext_vector_type(4)));
typedef int          v8i __attribute__((ext_vector_type(8)));
typedef int          v4i __attribute__((ext_vector_type(4)));

#if defined(__has_builtin)
#if __has_builtin(__builtin_amdgcn_tensor_load_to_lds)
#define HAVE_TDM 1
#endif
#endif
#ifndef HAVE_TDM
#define HAVE_TDM 0
#endif

// ---- TDM: 2D tile (rows x d0 elems, bf16) global -> LDS --------------------
// Descriptor per ISA cdna5 §8.3/8.4: group0 = {flags, lds_addr, gaddr, type},
// group1 = {mask/data_size, dims, tile dims, dim0 stride}.
#if HAVE_TDM
__device__ __forceinline__ void tdm_load_2d_bf16(unsigned lds_off,
                                                 const void* gptr,
                                                 unsigned d0_elems,
                                                 unsigned rows,
                                                 unsigned stride_elems) {
  unsigned long long ga = (unsigned long long)(size_t)gptr;
  v4u g0;
  g0[0] = 1u;                                       // count=1, user descriptor
  g0[1] = lds_off;                                  // lds_addr (bytes)
  g0[2] = (unsigned)(ga & 0xFFFFFFFFu);             // global_addr lo
  g0[3] = (unsigned)((ga >> 32) & 0x1FFFFFFu) | (2u << 30);  // addr hi | type=2
  v8i g1;
  g1[0] = (int)(1u << 16);                          // wg_mask=0, data_size=2B
  g1[1] = (int)((d0_elems & 0xFFFFu) << 16);        // tensor_dim0 lo
  g1[2] = (int)(((d0_elems >> 16) & 0xFFFFu) | ((rows & 0xFFFFu) << 16));
  g1[3] = (int)(((rows >> 16) & 0xFFFFu) | ((d0_elems & 0xFFFFu) << 16)); // tile_dim0
  g1[4] = (int)(rows & 0xFFFFu);                    // tile_dim1 | tile_dim2=0
  g1[5] = (int)stride_elems;                        // tensor_dim0_stride lo32
  g1[6] = 0;
  g1[7] = 0;
  v4i z4 = {0, 0, 0, 0};
#if __clang_major__ >= 23
  v8i z8 = {0, 0, 0, 0, 0, 0, 0, 0};
  __builtin_amdgcn_tensor_load_to_lds(g0, g1, z4, z4, z8, 0);
#else
  __builtin_amdgcn_tensor_load_to_lds(g0, g1, z4, z4, 0);
#endif
#if __has_builtin(__builtin_amdgcn_s_wait_tensorcnt)
  __builtin_amdgcn_s_wait_tensorcnt(0);
#else
  asm volatile("s_wait_tensorcnt 0x0" ::: "memory");
#endif
}
#endif

// Stage a (rows x d0) bf16 tile into LDS; wave 0 drives the TDM, then barrier.
__device__ __forceinline__ void stage_tile(__bf16* lds, const __bf16* g,
                                           int rows, int d0, int stride) {
#if HAVE_TDM
  if (threadIdx.x < 32) {
    tdm_load_2d_bf16((unsigned)(size_t)(void*)lds, (const void*)g,
                     (unsigned)d0, (unsigned)rows, (unsigned)stride);
  }
#else
  int n8 = rows * d0 / 8;
  for (int i = threadIdx.x; i < n8; i += blockDim.x) {
    int r = (i * 8) / d0, c = (i * 8) % d0;
    *reinterpret_cast<v8bf*>(lds + r * d0 + c) =
        *reinterpret_cast<const v8bf*>(g + (size_t)r * stride + c);
  }
#endif
  __syncthreads();
}

// ---- fragment helpers (ISA 7.12.2 wave32 16-bit A/B layout) ----------------
// Per lane: row = lane&15, k-halves by lane>>4:
// elems 0..7 <- p[0..7] (K=kb+0..7), elems 8..15 <- p[16..23] (K=kb+16..23)
__device__ __forceinline__ v16bf load_frag(const __bf16* __restrict__ p) {
  v8bf lo = *reinterpret_cast<const v8bf*>(p);
  v8bf hi = *reinterpret_cast<const v8bf*>(p + 16);
  v16bf f;
#pragma unroll
  for (int i = 0; i < 8; ++i) { f[i] = lo[i]; f[i + 8] = hi[i]; }
  return f;
}

__device__ __forceinline__ v16bf zero_frag() {
  v16bf f;
#pragma unroll
  for (int i = 0; i < 16; ++i) f[i] = (__bf16)0.0f;
  return f;
}

__device__ __forceinline__ float sigmoidf_(float x) {
  return 1.0f / (1.0f + __expf(-x));
}

// ---- elementwise fp32 -> bf16 ---------------------------------------------
__global__ void k_f32_to_bf16(const float* __restrict__ src,
                              __bf16* __restrict__ dst, long long n) {
  long long i = (long long)blockIdx.x * blockDim.x + threadIdx.x;
  long long stride = (long long)gridDim.x * blockDim.x;
  for (; i < n; i += stride) dst[i] = (__bf16)src[i];
}

// ---- embedding gather + convert -------------------------------------------
__global__ void k_embed_gather(const int* __restrict__ x,
                               const float* __restrict__ embW,
                               __bf16* __restrict__ emb) {
  int row = blockIdx.x;               // row = b*T + t
  int tok = x[row];
  const float* src = embW + (size_t)tok * EMB;
  __bf16* dst = emb + (size_t)row * EMB;
  for (int e = threadIdx.x; e < EMB; e += blockDim.x) dst[e] = (__bf16)src[e];
}

// ---- zero h0 (bf16) and c0 (f32) ------------------------------------------
__global__ void k_init_state(__bf16* __restrict__ h, float* __restrict__ c) {
  int i = blockIdx.x * blockDim.x + threadIdx.x;
  if (i < BATCH * HID) { h[i] = (__bf16)0.0f; c[i] = 0.0f; }
}

// ---- x_proj = emb @ W_ih^T + (b_ih + b_hh) --------------------------------
// Block = 256 thr = 8 waves as 2(M) x 4(N); wave tile 16 x 64.
// A block-tile (32 x 1024, 64 KB) TDM-staged into LDS, reused by all waves.
__global__ void k_gemm_xproj(const __bf16* __restrict__ A,
                             const __bf16* __restrict__ Bm,
                             const float* __restrict__ b_ih,
                             const float* __restrict__ b_hh,
                             float* __restrict__ C) {
  const int K = EMB;
  __shared__ __bf16 tileA[32 * EMB];

  stage_tile(tileA, A + (size_t)blockIdx.x * 32 * K, 32, K, K);

  int lane  = threadIdx.x & 31;
  int wave  = threadIdx.x >> 5;
  int wm    = wave >> 2;               // 0..1
  int wn    = wave & 3;                // 0..3
  int m0    = blockIdx.x * 32 + wm * 16;
  int n0    = blockIdx.y * 256 + wn * 64;
  int mrow  = lane & 15;
  int kb    = (lane >> 4) * 8;

  v8f acc[4] = {};
  const __bf16* aptr = tileA + (wm * 16 + mrow) * K + kb;   // LDS (ds_load)

  for (int k = 0; k < K; k += 32) {
    if (k + 32 < K)
      __builtin_prefetch(Bm + (size_t)(n0 + mrow) * K + kb + k + 32, 0, 1);
    v16bf bfrag[4];
#pragma unroll
    for (int j = 0; j < 4; ++j)
      bfrag[j] = load_frag(Bm + (size_t)(n0 + j * 16 + mrow) * K + kb + k);
    v16bf af = load_frag(aptr + k);
#pragma unroll
    for (int j = 0; j < 4; ++j)
      acc[j] = __builtin_amdgcn_wmma_f32_16x16x32_bf16(
          false, af, false, bfrag[j], (short)0, acc[j], false, false);
  }

  int lhi = lane >> 4;
#pragma unroll
  for (int j = 0; j < 4; ++j) {
    int n = n0 + j * 16 + (lane & 15);
    float bias = b_ih[n] + b_hh[n];
#pragma unroll
    for (int r = 0; r < 8; ++r) {
      int m = m0 + r + 8 * lhi;
      C[(size_t)m * G4H + n] = acc[j][r] + bias;
    }
  }
}

// ---- recurrent step, matrix part: gates = x_proj[:,t,:] + h @ W_hh^T ------
// M = 8 (padded to 16 via zero lanes), N = 4096, K = 1024.
// h (8 x 1024, 16 KB) TDM-staged into LDS once per block.
__global__ void k_lstm_matmul(const __bf16* __restrict__ h,
                              const __bf16* __restrict__ Whh,
                              const float* __restrict__ xproj,
                              float* __restrict__ gates, int t) {
  const int K = HID;
  __shared__ __bf16 tileH[BATCH * HID];

  stage_tile(tileH, h, BATCH, K, K);

  int lane = threadIdx.x & 31;
  int wave = threadIdx.x >> 5;
  int n0   = (blockIdx.x * 8 + wave) * 16;
  int mrow = lane & 15;
  int kb   = (lane >> 4) * 8;
  bool arow_valid = (mrow < BATCH);

  v8f acc = {};
  const __bf16* aptr = tileH + mrow * K + kb;               // LDS (ds_load)
  const __bf16* bptr = Whh + (size_t)(n0 + mrow) * K + kb;

  for (int k = 0; k < K; k += 32) {
    v16bf bf = load_frag(bptr + k);
    v16bf af = arow_valid ? load_frag(aptr + k) : zero_frag();
    acc = __builtin_amdgcn_wmma_f32_16x16x32_bf16(
        false, af, false, bf, (short)0, acc, false, false);
  }

  if ((lane >> 4) == 0) {            // lanes 0..15 hold M = 0..7 in r = 0..7
    int n = n0 + (lane & 15);
#pragma unroll
    for (int r = 0; r < BATCH; ++r) {
      gates[(size_t)r * G4H + n] =
          acc[r] + xproj[((size_t)r * SEQT + t) * G4H + n];
    }
  }
}

// ---- recurrent step, pointwise gates --------------------------------------
__global__ void k_lstm_point(const float* __restrict__ gates,
                             float* __restrict__ c,
                             float* __restrict__ h_f32,
                             __bf16* __restrict__ h_bf16,
                             __bf16* __restrict__ out_bf16, int t) {
  int i = blockIdx.x * blockDim.x + threadIdx.x;      // 0..8191
  if (i >= BATCH * HID) return;
  int b = i >> 10, j = i & 1023;
  const float* g = gates + (size_t)b * G4H;
  float ig = sigmoidf_(g[j]);
  float fg = sigmoidf_(g[HID + j]);
  float gg = tanhf(g[2 * HID + j]);
  float og = sigmoidf_(g[3 * HID + j]);
  float cn = fg * c[i] + ig * gg;
  float hn = og * tanhf(cn);
  c[i] = cn;
  h_f32[i] = hn;
  h_bf16[i] = (__bf16)hn;
  out_bf16[((size_t)b * SEQT + t) * HID + j] = (__bf16)hn;
}

// ---- decoder: logits = out @ emb_W^T + dec_b ------------------------------
// A block-tile (32 x 1024) TDM-staged; B rows (emb_W bf16, 103 MB) stream
// through L2 (fits the 192 MB global L2). N-edge guarded for 50257.
__global__ void k_gemm_decode(const __bf16* __restrict__ A,
                              const __bf16* __restrict__ Bm,
                              const float* __restrict__ dec_b,
                              float* __restrict__ C) {
  const int K = HID;
  __shared__ __bf16 tileA[32 * HID];

  stage_tile(tileA, A + (size_t)blockIdx.x * 32 * K, 32, K, K);

  int lane = threadIdx.x & 31;
  int wave = threadIdx.x >> 5;
  int wm   = wave >> 2;
  int wn   = wave & 3;
  int m0   = blockIdx.x * 32 + wm * 16;
  int n0   = blockIdx.y * 256 + wn * 64;
  int mrow = lane & 15;
  int kb   = (lane >> 4) * 8;

  v8f acc[4] = {};
  const __bf16* aptr = tileA + (wm * 16 + mrow) * K + kb;   // LDS (ds_load)
  bool nvalid[4];
#pragma unroll
  for (int j = 0; j < 4; ++j) nvalid[j] = (n0 + j * 16 + mrow) < VOCAB;

  for (int k = 0; k < K; k += 32) {
    if (k + 32 < K && nvalid[0])
      __builtin_prefetch(Bm + (size_t)(n0 + mrow) * K + kb + k + 32, 0, 1);
    v16bf bfrag[4];
#pragma unroll
    for (int j = 0; j < 4; ++j)
      bfrag[j] = nvalid[j]
          ? load_frag(Bm + (size_t)(n0 + j * 16 + mrow) * K + kb + k)
          : zero_frag();
    v16bf af = load_frag(aptr + k);
#pragma unroll
    for (int j = 0; j < 4; ++j)
      acc[j] = __builtin_amdgcn_wmma_f32_16x16x32_bf16(
          false, af, false, bfrag[j], (short)0, acc[j], false, false);
  }

  int lhi = lane >> 4;
#pragma unroll
  for (int j = 0; j < 4; ++j) {
    int n = n0 + j * 16 + (lane & 15);
    if (n < VOCAB) {
      float bias = dec_b[n];
#pragma unroll
      for (int r = 0; r < 8; ++r) {
        int m = m0 + r + 8 * lhi;
        C[(size_t)m * VOCAB + n] = acc[j][r] + bias;
      }
    }
  }
}

// ---- per-row online log-sum-exp -------------------------------------------
__global__ void k_lse(const float* __restrict__ logits, float* __restrict__ lse) {
  int row = blockIdx.x;
  const float* p = logits + (size_t)row * VOCAB;
  float m = -INFINITY, s = 0.0f;
  for (int i = threadIdx.x; i < VOCAB; i += blockDim.x) {
    float v = p[i];
    if (v > m) { s = s * __expf(m - v) + 1.0f; m = v; }
    else       { s += __expf(v - m); }
  }
  __shared__ float sm[256];
  __shared__ float ss[256];
  sm[threadIdx.x] = m; ss[threadIdx.x] = s;
  __syncthreads();
  for (int off = 128; off > 0; off >>= 1) {
    if ((int)threadIdx.x < off) {
      float m2 = sm[threadIdx.x + off], s2 = ss[threadIdx.x + off];
      float M = fmaxf(sm[threadIdx.x], m2);
      ss[threadIdx.x] = ss[threadIdx.x] * __expf(sm[threadIdx.x] - M) +
                        s2 * __expf(m2 - M);
      sm[threadIdx.x] = M;
    }
    __syncthreads();
  }
  if (threadIdx.x == 0) lse[row] = sm[0] + __logf(ss[0]);
}

__global__ void k_sub_lse(float* __restrict__ logits,
                          const float* __restrict__ lse) {
  int row = blockIdx.x;
  float l = lse[row];
  float* p = logits + (size_t)row * VOCAB;
  for (int i = threadIdx.x; i < VOCAB; i += blockDim.x) p[i] -= l;
}

// ---- hidden tail (hT, cT) -------------------------------------------------
__global__ void k_tail(float* __restrict__ out,
                       const float* __restrict__ h,
                       const float* __restrict__ c) {
  int i = blockIdx.x * blockDim.x + threadIdx.x;
  size_t base = (size_t)BT * VOCAB;
  if (i < BATCH * HID) {
    out[base + i] = h[i];
    out[base + BATCH * HID + i] = c[i];
  }
}

// ---------------------------------------------------------------------------
extern "C" void kernel_launch(void* const* d_in, const int* in_sizes, int n_in,
                              void* d_out, int out_size, void* d_ws,
                              size_t ws_size, hipStream_t stream) {
  (void)in_sizes; (void)n_in; (void)out_size; (void)ws_size;

  const int*   x      = (const int*)  d_in[0];
  const float* emb_W  = (const float*)d_in[1];
  const float* W_ih   = (const float*)d_in[2];
  const float* W_hh   = (const float*)d_in[3];
  const float* b_ih   = (const float*)d_in[4];
  const float* b_hh   = (const float*)d_in[5];
  const float* dec_b  = (const float*)d_in[6];
  float*       out    = (float*)d_out;

  // ---- workspace layout (256B-aligned chunks) ----
  char* ws = (char*)d_ws;
  size_t off = 0;
  auto take = [&](size_t bytes) { char* p = ws + off; off += (bytes + 255) & ~(size_t)255; return p; };

  __bf16* embW_bf = (__bf16*)take((size_t)VOCAB * EMB * 2);   // 103 MB (fits L2)
  __bf16* Wih_bf  = (__bf16*)take((size_t)G4H * EMB * 2);     //   8 MB
  __bf16* Whh_bf  = (__bf16*)take((size_t)G4H * HID * 2);     //   8 MB (L2-resident)
  __bf16* emb_bf  = (__bf16*)take((size_t)BT * EMB * 2);      //   4 MB
  __bf16* out_bf  = (__bf16*)take((size_t)BT * HID * 2);      //   4 MB
  float*  xproj   = (float*) take((size_t)BT * G4H * 4);      //  32 MB
  float*  gates   = (float*) take((size_t)BATCH * G4H * 4);
  __bf16* h_bf    = (__bf16*)take((size_t)BATCH * HID * 2);
  float*  h_f32   = (float*) take((size_t)BATCH * HID * 4);
  float*  c_f32   = (float*) take((size_t)BATCH * HID * 4);
  float*  lse     = (float*) take((size_t)BT * 4);

  // ---- 1. weight conversions to bf16 ----
  k_f32_to_bf16<<<4096, 256, 0, stream>>>(emb_W, embW_bf, (long long)VOCAB * EMB);
  k_f32_to_bf16<<<1024, 256, 0, stream>>>(W_ih, Wih_bf, (long long)G4H * EMB);
  k_f32_to_bf16<<<1024, 256, 0, stream>>>(W_hh, Whh_bf, (long long)G4H * HID);

  // ---- 2. embedding gather (+convert) ----
  k_embed_gather<<<BT, 256, 0, stream>>>(x, emb_W, emb_bf);

  // ---- 3. state init ----
  k_init_state<<<32, 256, 0, stream>>>(h_bf, c_f32);

  // ---- 4. input projection GEMM (WMMA bf16, TDM-staged A) ----
  {
    dim3 grid(BT / 32, G4H / 256);       // (64, 16)
    k_gemm_xproj<<<grid, 256, 0, stream>>>(emb_bf, Wih_bf, b_ih, b_hh, xproj);
  }

  // ---- 5. 256 sequential LSTM steps ----
  for (int t = 0; t < SEQT; ++t) {
    k_lstm_matmul<<<G4H / 128, 256, 0, stream>>>(h_bf, Whh_bf, xproj, gates, t);
    k_lstm_point<<<32, 256, 0, stream>>>(gates, c_f32, h_f32, h_bf, out_bf, t);
  }

  // ---- 6. tied decoder GEMM (WMMA bf16, TDM-staged A) -> logits ----
  {
    dim3 grid(BT / 32, (VOCAB + 255) / 256);   // (64, 197)
    k_gemm_decode<<<grid, 256, 0, stream>>>(out_bf, embW_bf, dec_b, out);
  }

  // ---- 7. log_softmax over VOCAB per row ----
  k_lse<<<BT, 256, 0, stream>>>(out, lse);
  k_sub_lse<<<BT, 256, 0, stream>>>(out, lse);

  // ---- 8. hidden state tail (hT, cT) ----
  k_tail<<<32, 256, 0, stream>>>(out, h_f32, c_f32);
}
